// OptimizedMOE_36197984371396
// MI455X (gfx1250) — matrix operand order repaired and checked by reading, not compiled, not run on hardware
//
#include <hip/hip_runtime.h>

// MoE conv block for MI455X (gfx1250, wave32, WMMA bf16 16x16x32).
// Pipeline: GAP+router -> x transpose->bf16 -> weight cvt -> shared 1x1 WMMA GEMM
//  -> top-2 expert expand 1x1 WMMA GEMM -> 3x3 implicit-GEMM WMMA conv + gated combine.
// The 3x3 conv stages its weight tiles into LDS with the Tensor Data Mover
// (tensor_load_to_lds, double-buffered, s_wait_tensorcnt + barrier) and software-
// pipelines the activation B-fragments so global latency hides behind the WMMAs.
// Workspace requirement: ~171 MiB (see offsets below).

#define BATCH 16
#define CIN   256
#define CHID  512
#define COUT  256
#define HH    64
#define WWI   64
#define HW    4096
#define NEXP  4

typedef __bf16 bf16_t;
typedef __attribute__((ext_vector_type(16))) __bf16 bf16x16;
typedef __attribute__((ext_vector_type(8)))  float  f32x8;

union FragU { bf16x16 v; uint4 q[2]; };

// ---- workspace layout (bytes) ----
#define OFF_GAP   ((size_t)0)            // 16*256*4       = 16384
#define OFF_TOPI  ((size_t)16384)        // 32 ints        = 128
#define OFF_WGT   ((size_t)16512)        // 32 floats      = 128
#define OFF_XB    ((size_t)16640)        // 16*4096*256*2  = 33554432  (x bf16, [b][hw][c])
#define OFF_WSB   ((size_t)33571072)     // 256*256*2      = 131072    (shared_w bf16)
#define OFF_W1B   ((size_t)33702144)     // 4*512*256*2    = 1048576   (ew1 bf16)
#define OFF_W2B   ((size_t)34750720)     // 4*9*256*512*2  = 9437184   (ew2 bf16 [e][ky][kx][o][c])
#define OFF_H     ((size_t)44187904)     // 16*2*4096*512*2= 134217728 (h bf16, [b][slot][hw][c])

// ---- TDM availability ----
#if defined(__has_builtin)
#if __has_builtin(__builtin_amdgcn_tensor_load_to_lds) && __has_builtin(__builtin_amdgcn_s_wait_tensorcnt)
#define USE_TDM 1
#endif
#endif
#ifndef USE_TDM
#define USE_TDM 0
#endif

__device__ __forceinline__ float silu_f(float x) { return x / (1.0f + __expf(-x)); }

__device__ __forceinline__ bf16x16 zero_frag() {
  FragU f; f.q[0] = make_uint4(0,0,0,0); f.q[1] = make_uint4(0,0,0,0); return f.v;
}

// per-lane A fragment (weights, M rows): lane L holds M=L&15,
// K = k0+16*(L>>4) .. +15 (16 contiguous bf16, two 16B chunks)
__device__ __forceinline__ bf16x16 load_a(const bf16_t* __restrict__ row, int ksel) {
  FragU a;
  a.q[0] = *(const uint4*)(row + 16 * ksel);
  a.q[1] = *(const uint4*)(row + 16 * ksel + 8);
  return a.v;
}

// ---------------- router ----------------
__global__ void k_gap(const float* __restrict__ x, float* __restrict__ gap) {
  __shared__ float red[128];
  int bc = blockIdx.x; // b*CIN + c
  const float* p = x + (size_t)bc * HW;
  float s = 0.f;
  for (int i = threadIdx.x; i < HW; i += 128) s += p[i];
  red[threadIdx.x] = s;
  __syncthreads();
  for (int off = 64; off > 0; off >>= 1) {
    if ((int)threadIdx.x < off) red[threadIdx.x] += red[threadIdx.x + off];
    __syncthreads();
  }
  if (threadIdx.x == 0) gap[bc] = red[0] * (1.0f / (float)HW);
}

__global__ void k_router(const float* __restrict__ gap, const float* __restrict__ rw,
                         int* __restrict__ topi, float* __restrict__ wgt) {
  int b = threadIdx.x;
  if (b >= BATCH) return;
  float lg[NEXP];
  for (int e = 0; e < NEXP; ++e) {
    float s = 0.f;
    for (int c = 0; c < CIN; ++c) s += gap[b * CIN + c] * rw[e * CIN + c];
    lg[e] = s;
  }
  int i0 = 0;
  for (int e = 1; e < NEXP; ++e) if (lg[e] > lg[i0]) i0 = e;
  int i1 = (i0 == 0) ? 1 : 0;
  for (int e = 0; e < NEXP; ++e) if (e != i0 && lg[e] > lg[i1]) i1 = e;
  // softmax-then-renormalize over top-2 == softmax over the two top logits
  float m = fmaxf(lg[i0], lg[i1]);
  float e0 = __expf(lg[i0] - m), e1 = __expf(lg[i1] - m);
  float inv = 1.0f / (e0 + e1);
  topi[b * 2 + 0] = i0; topi[b * 2 + 1] = i1;
  wgt[b * 2 + 0] = e0 * inv; wgt[b * 2 + 1] = e1 * inv;
}

// ---------------- data prep ----------------
// x [b][c][hw] f32 -> xb [b][hw][c] bf16, LDS-tiled transpose
__global__ __launch_bounds__(256) void k_transpose_x(const float* __restrict__ x,
                                                     bf16_t* __restrict__ xb) {
  __shared__ float tile[32][33];
  int c0 = blockIdx.x * 32, p0 = blockIdx.y * 32, b = blockIdx.z;
  int tx = threadIdx.x, ty = threadIdx.y;
  const float* src = x + (size_t)b * CIN * HW;
  #pragma unroll
  for (int i = 0; i < 4; ++i) {
    int c = ty + i * 8;
    tile[c][tx] = src[(size_t)(c0 + c) * HW + p0 + tx];
  }
  __syncthreads();
  bf16_t* dst = xb + (size_t)b * HW * CIN;
  #pragma unroll
  for (int i = 0; i < 4; ++i) {
    int p = ty + i * 8;
    dst[(size_t)(p0 + p) * CIN + c0 + tx] = (bf16_t)tile[tx][p];
  }
}

__global__ void k_cvt(const float* __restrict__ s, bf16_t* __restrict__ d, int n) {
  int i = blockIdx.x * 256 + threadIdx.x;
  if (i < n) d[i] = (bf16_t)s[i];
}

// ew2 [e][o][c][ky][kx] f32 -> w2b [e][ky][kx][o][c] bf16
__global__ void k_cvt_w2(const float* __restrict__ s, bf16_t* __restrict__ d) {
  int i = blockIdx.x * 256 + threadIdx.x;
  if (i >= NEXP * 9 * COUT * CHID) return;
  int c = i % CHID;
  int t = i / CHID;
  int o = t % COUT; t /= COUT;
  int kk = t % 9;  int e = t / 9;
  int ky = kk / 3, kx = kk % 3;
  d[i] = (bf16_t)s[((((size_t)e * COUT + o) * CHID + c) * 3 + ky) * 3 + kx];
}

// ---------------- shared expert: 1x1 GEMM + BN + SiLU -> d_out ----------------
__global__ __launch_bounds__(256) void k_gemm_shared(
    const bf16_t* __restrict__ xb, const bf16_t* __restrict__ wsb,
    const float* __restrict__ gamma, const float* __restrict__ beta,
    float* __restrict__ out) {
  int b = blockIdx.y;
  int tid = threadIdx.x, lane = tid & 31, w = tid >> 5;
  int nl = lane & 15, ksel = lane >> 4;
  int m0 = (w & 3) * 64;                       // 4 waves cover M=256
  int n0 = blockIdx.x * 32 + (w >> 2) * 16;    // 2 N-tiles per block
  const bf16_t* bbase = xb + ((size_t)b * HW + n0 + nl) * CIN;
  f32x8 acc[4];
  #pragma unroll
  for (int mi = 0; mi < 4; ++mi) for (int r = 0; r < 8; ++r) acc[mi][r] = 0.f;
  for (int k0 = 0; k0 < CIN; k0 += 32) {
    bf16x16 bf = *(const bf16x16*)(bbase + k0 + 16 * ksel);
    #pragma unroll
    for (int mi = 0; mi < 4; ++mi) {
      bf16x16 a = load_a(wsb + (size_t)(m0 + mi * 16 + nl) * CIN + k0, ksel);
      acc[mi] = __builtin_amdgcn_wmma_f32_16x16x32_bf16(false, a, false, bf,
                                                        (short)0, acc[mi], false, false);
    }
  }
  int n = n0 + nl;
  #pragma unroll
  for (int mi = 0; mi < 4; ++mi)
    #pragma unroll
    for (int r = 0; r < 8; ++r) {
      int m = m0 + mi * 16 + 8 * ksel + r;
      float v = acc[mi][r] * gamma[m] + beta[m];
      out[((size_t)b * COUT + m) * HW + n] = silu_f(v);
    }
}

// ---------------- top-2 expert expand: 1x1 GEMM + BN + SiLU -> h bf16 ----------------
__global__ __launch_bounds__(256) void k_gemm_expand(
    const bf16_t* __restrict__ xb, const bf16_t* __restrict__ w1b,
    const float* __restrict__ eg1, const float* __restrict__ eb1,
    const int* __restrict__ topi, bf16_t* __restrict__ h) {
  int b = blockIdx.z, slot = blockIdx.y;
  int e = topi[b * 2 + slot];
  int tid = threadIdx.x, lane = tid & 31, w = tid >> 5;
  int nl = lane & 15, ksel = lane >> 4;
  int m0 = w * 64;                 // 8 waves cover M=512
  int n0 = blockIdx.x * 16;
  const bf16_t* bbase = xb + ((size_t)b * HW + n0 + nl) * CIN;
  const bf16_t* wbase = w1b + (size_t)e * CHID * CIN;
  f32x8 acc[4];
  #pragma unroll
  for (int mi = 0; mi < 4; ++mi) for (int r = 0; r < 8; ++r) acc[mi][r] = 0.f;
  for (int k0 = 0; k0 < CIN; k0 += 32) {
    bf16x16 bf = *(const bf16x16*)(bbase + k0 + 16 * ksel);
    #pragma unroll
    for (int mi = 0; mi < 4; ++mi) {
      bf16x16 a = load_a(wbase + (size_t)(m0 + mi * 16 + nl) * CIN + k0, ksel);
      acc[mi] = __builtin_amdgcn_wmma_f32_16x16x32_bf16(false, a, false, bf,
                                                        (short)0, acc[mi], false, false);
    }
  }
  int n = n0 + nl;
  bf16_t* hb = h + ((size_t)(b * 2 + slot) * HW + n) * CHID;
  #pragma unroll
  for (int mi = 0; mi < 4; ++mi) {
    union { bf16_t hv[8]; uint4 q; } u;
    int mbase = m0 + mi * 16 + 8 * ksel;   // 8 consecutive M per lane -> one b128 store
    #pragma unroll
    for (int r = 0; r < 8; ++r) {
      int m = mbase + r;
      float v = acc[mi][r] * eg1[e * CHID + m] + eb1[e * CHID + m];
      u.hv[r] = (bf16_t)silu_f(v);
    }
    *(uint4*)(hb + mbase) = u.q;
  }
}

// ---------------- 3x3 conv weight staging ----------------
// Weight chunk per step: [COUT=256 rows x CW=32 c] from w2b[e][ky][kx], row stride CHID.
// LDS layout: row stride 80B (64B data + 16B pad) so 16 consecutive M rows map to
// distinct bank groups for the per-lane ds_load_b128 fragment reads.
#define CW      32
#define NCHUNK  (CHID / CW)          // 16
#define NSTEPS  (9 * NCHUNK)         // 144
#define LDSROW  80
#define WBUFSZ  (COUT * LDSROW)      // 20480 B per buffer

#if USE_TDM
typedef __attribute__((ext_vector_type(4))) unsigned int uint32x4_t_;
typedef __attribute__((ext_vector_type(8))) int int32x8_t_;
typedef __attribute__((ext_vector_type(4))) int int32x4_t_;

__device__ __forceinline__ void tdm_issue_w2(const bf16_t* src, unsigned lds_off) {
  unsigned long long ga = (unsigned long long)(uintptr_t)src;
  uint32x4_t_ g0;
  g0[0] = 1u;                                  // count=1, user descriptor
  g0[1] = lds_off;                             // LDS byte address of dest buffer
  g0[2] = (unsigned)ga;                        // global_addr lo
  g0[3] = (unsigned)(ga >> 32) | (2u << 30);   // global_addr hi | type=2
  int32x8_t_ g1 = (int32x8_t_)0;
  // data_size=2B, pad_enable, pad_interval=16 DWORDs (enc 3), pad_amount=4 DWORDs (enc 3)
  g1[0] = (int)((1u << 16) | (1u << 20) | (3u << 22) | (3u << 25));
  g1[1] = (int)(512u << 16);                   // tensor_dim0 = 512 (lo16)
  g1[2] = (int)(256u << 16);                   // tensor_dim0 hi=0 | tensor_dim1 lo16 = 256
  g1[3] = (int)((unsigned)CW << 16);           // tensor_dim1 hi=0 | tile_dim0 = 32
  g1[4] = 256;                                 // tile_dim1 = 256 rows (tile_dim2 unused)
  g1[5] = 512;                                 // tensor_dim0_stride = 512 elements
  int32x4_t_ gz4 = (int32x4_t_)0;              // 2D tensor: groups 2/3 unused
  int32x8_t_ gz8 = (int32x8_t_)0;              // extra group (6-arg toolchain variant)
  __builtin_amdgcn_tensor_load_to_lds(g0, g1, gz4, gz4, gz8, 0);
}
#endif

__device__ __forceinline__ void copy_w2_chunk(const bf16_t* __restrict__ src,
                                              unsigned char* dst, int tid) {
  // fallback: 256 threads, thread t copies row t (64B) global -> LDS (padded rows)
  const uint4* s = (const uint4*)(src + (size_t)tid * CHID);
  uint4* d = (uint4*)(dst + tid * LDSROW);
  d[0] = s[0]; d[1] = s[1]; d[2] = s[2]; d[3] = s[3];
}

// B-fragment for conv step s: activation pixel (py+ky-1, px+kx-1), c-chunk (s&15)*CW,
// zero-filled outside the image (SAME padding).
__device__ __forceinline__ bf16x16 load_b_conv(const bf16_t* __restrict__ hb,
                                               int s, int py, int px, int ksel) {
  int tap = s >> 4;
  int ky = tap / 3, kx = tap % 3;
  int c0g = (s & 15) * CW;
  int sy = py + ky - 1, sx = px + kx - 1;
  bool ok = (sy >= 0) && (sy < HH) && (sx >= 0) && (sx < WWI);
  long pix = ok ? (long)(sy * WWI + sx) : 0;
  return ok ? *(const bf16x16*)(hb + pix * CHID + c0g + 16 * ksel) : zero_frag();
}

// ---------------- 3x3 conv (implicit GEMM, K=512*9) + BN + SiLU + gated combine ----------------
__global__ __launch_bounds__(256) void k_conv3(
    const bf16_t* __restrict__ h, const bf16_t* __restrict__ w2b,
    const float* __restrict__ eg2, const float* __restrict__ eb2,
    const int* __restrict__ topi, const float* __restrict__ wgt,
    float* __restrict__ out) {
  __shared__ unsigned char ldsw[2 * WBUFSZ];   // 40 KiB double buffer
  int b = blockIdx.z, py = blockIdx.y;
  int tid = threadIdx.x, lane = tid & 31, w = tid >> 5;
  int nl = lane & 15, ksel = lane >> 4;
  int m0 = (w & 3) * 64;
  int x0 = blockIdx.x * 32 + (w >> 2) * 16;
  int px = x0 + nl;
  int n = py * WWI + px;
#if USE_TDM
  unsigned lds_base = (unsigned)(unsigned long long)(uintptr_t)(void*)&ldsw[0];
#endif

  float accOut[4][8];
  #pragma unroll
  for (int mi = 0; mi < 4; ++mi)
    #pragma unroll
    for (int r = 0; r < 8; ++r)
      accOut[mi][r] = out[((size_t)b * COUT + (m0 + mi * 16 + 8 * ksel + r)) * HW + n];

  for (int slot = 0; slot < 2; ++slot) {
    int e = topi[b * 2 + slot];
    float wg = wgt[b * 2 + slot];
    const bf16_t* wslot = w2b + (size_t)e * 9 * COUT * CHID;
    const bf16_t* hb = h + (size_t)(b * 2 + slot) * HW * CHID;
    f32x8 acc[4];
    #pragma unroll
    for (int mi = 0; mi < 4; ++mi) for (int r = 0; r < 8; ++r) acc[mi][r] = 0.f;

    // prologue: stage weight chunk 0 into buffer 0; prefetch B-fragment for step 0
#if USE_TDM
    if (w == 0) tdm_issue_w2(wslot, lds_base);
#else
    copy_w2_chunk(wslot, &ldsw[0], tid);
#endif
    bf16x16 bfcur = load_b_conv(hb, 0, py, px, ksel);

    for (int s = 0; s < NSTEPS; ++s) {
      int bufi = s & 1;
#if USE_TDM
      if (w == 0) __builtin_amdgcn_s_wait_tensorcnt(0);  // weight chunk s landed
#endif
      __syncthreads();  // chunk s visible; everyone done reading buf[bufi^1]
      if (s + 1 < NSTEPS) {
        int s1 = s + 1;
        const bf16_t* src = wslot + (size_t)(s1 >> 4) * COUT * CHID + (size_t)(s1 & 15) * CW;
#if USE_TDM
        if (w == 0) tdm_issue_w2(src, lds_base + (unsigned)((bufi ^ 1) * WBUFSZ));
#else
        copy_w2_chunk(src, &ldsw[(bufi ^ 1) * WBUFSZ], tid);
#endif
      }
      // prefetch next B-fragment; its load latency hides behind this step's WMMAs
      bf16x16 bfnext = (s + 1 < NSTEPS) ? load_b_conv(hb, s + 1, py, px, ksel) : zero_frag();
      // compute step s: A-fragments from LDS (TDM-staged), B-fragment prefetched
      const unsigned char* bufp = &ldsw[bufi * WBUFSZ];
      #pragma unroll
      for (int mi = 0; mi < 4; ++mi) {
        const unsigned char* rowp = bufp + (m0 + mi * 16 + nl) * LDSROW + 32 * ksel;
        FragU a;
        a.q[0] = *(const uint4*)(rowp);
        a.q[1] = *(const uint4*)(rowp + 16);
        acc[mi] = __builtin_amdgcn_wmma_f32_16x16x32_bf16(false, a.v, false, bfcur,
                                                          (short)0, acc[mi], false, false);
      }
      bfcur = bfnext;
    }
    #pragma unroll
    for (int mi = 0; mi < 4; ++mi)
      #pragma unroll
      for (int r = 0; r < 8; ++r) {
        int m = m0 + mi * 16 + 8 * ksel + r;
        float v = acc[mi][r] * eg2[e * COUT + m] + eb2[e * COUT + m];
        accOut[mi][r] += wg * silu_f(v);
      }
    __syncthreads();  // slot boundary: buffers reused by next slot's prologue
  }
  #pragma unroll
  for (int mi = 0; mi < 4; ++mi)
    #pragma unroll
    for (int r = 0; r < 8; ++r)
      out[((size_t)b * COUT + (m0 + mi * 16 + 8 * ksel + r)) * HW + n] = accOut[mi][r];
}

extern "C" void kernel_launch(void* const* d_in, const int* in_sizes, int n_in,
                              void* d_out, int out_size, void* d_ws, size_t ws_size,
                              hipStream_t stream) {
  (void)in_sizes; (void)n_in; (void)out_size; (void)ws_size; // needs ~171 MiB of ws
  const float* x   = (const float*)d_in[0];
  const float* rw  = (const float*)d_in[1];
  const float* shw = (const float*)d_in[2];
  const float* shg = (const float*)d_in[3];
  const float* shb = (const float*)d_in[4];
  const float* ew1 = (const float*)d_in[5];
  const float* eg1 = (const float*)d_in[6];
  const float* eb1 = (const float*)d_in[7];
  const float* ew2 = (const float*)d_in[8];
  const float* eg2 = (const float*)d_in[9];
  const float* eb2 = (const float*)d_in[10];
  float* out = (float*)d_out;
  char* ws = (char*)d_ws;

  float*  gap  = (float*)(ws + OFF_GAP);
  int*    topi = (int*)(ws + OFF_TOPI);
  float*  wgt  = (float*)(ws + OFF_WGT);
  bf16_t* xb   = (bf16_t*)(ws + OFF_XB);
  bf16_t* wsb  = (bf16_t*)(ws + OFF_WSB);
  bf16_t* w1b  = (bf16_t*)(ws + OFF_W1B);
  bf16_t* w2b  = (bf16_t*)(ws + OFF_W2B);
  bf16_t* hbuf = (bf16_t*)(ws + OFF_H);

  k_gap<<<BATCH * CIN, 128, 0, stream>>>(x, gap);
  k_router<<<1, 32, 0, stream>>>(gap, rw, topi, wgt);
  k_transpose_x<<<dim3(CIN / 32, HW / 32, BATCH), dim3(32, 8), 0, stream>>>(x, xb);
  k_cvt<<<(COUT * CIN + 255) / 256, 256, 0, stream>>>(shw, wsb, COUT * CIN);
  k_cvt<<<(NEXP * CHID * CIN + 255) / 256, 256, 0, stream>>>(ew1, w1b, NEXP * CHID * CIN);
  k_cvt_w2<<<(NEXP * 9 * COUT * CHID + 255) / 256, 256, 0, stream>>>(ew2, w2b);
  k_gemm_shared<<<dim3(HW / 32, BATCH), 256, 0, stream>>>(xb, wsb, shg, shb, out);
  k_gemm_expand<<<dim3(HW / 16, 2, BATCH), 256, 0, stream>>>(xb, w1b, eg1, eb1, topi, hbuf);
  k_conv3<<<dim3(WWI / 32, HH, BATCH), 256, 0, stream>>>(hbuf, w2b, eg2, eb2, topi, wgt, out);
}
